// Mamba2Block_57346403336599
// MI455X (gfx1250) — matrix operand order repaired and checked by reading, not compiled, not run on hardware
//
#include <hip/hip_runtime.h>

typedef unsigned short u16;
typedef unsigned int   u32;
typedef unsigned long long u64;

typedef __attribute__((ext_vector_type(16))) __bf16 v16bf;
typedef __attribute__((ext_vector_type(8)))  float  v8f;
typedef __attribute__((ext_vector_type(4)))  unsigned int v4u;
typedef __attribute__((ext_vector_type(8)))  int v8i;
typedef __attribute__((ext_vector_type(4)))  int v4i;

union FragAB { v16bf v; u32 u[8]; };
union Pk8  { uint4 q; u16 h[8]; u32 w[4]; };
union Pk4f { float4 q; float f[4]; };

#if __has_builtin(__builtin_amdgcn_tensor_load_to_lds)
#define HAVE_TDM 1
#if __has_include(<hip/amd_detail/amd_gfx1250_TDM.h>)
#define TDM_6ARG 1
#endif
#else
#define HAVE_TDM 0
#endif

enum : int {
  NB    = 2,
  SEQL  = 4096,
  TOK   = NB * SEQL,                 // 8192 tokens
  DM    = 2048,
  DIN   = 4096,
  DST   = 64,
  NH    = 32,
  HD    = 128,
  CONVD = DIN + 2 * DST,             // 4224
  NPROJ = 2 * DIN + 2 * DST + NH,    // 8352
  CLEN  = 64,
  NCH   = SEQL / CLEN,               // 64 chunks per batch
  CHS   = NH * HD * DST,             // 262144 floats per (b,chunk) state
};

// ---------- bf16 helpers (RNE round) ----------
__device__ __forceinline__ u16 f2bf(float f) {
  u32 u = __float_as_uint(f);
  u32 r = (u + 0x7FFFu + ((u >> 16) & 1u)) >> 16;
  return (u16)r;
}
__device__ __forceinline__ float bf2f(u16 h) {
  return __uint_as_float(((u32)h) << 16);
}
__device__ __forceinline__ float siluf(float v) { return v / (1.0f + __expf(-v)); }

__device__ __forceinline__ v8f vzero8() {
  v8f z;
#pragma unroll
  for (int e = 0; e < 8; ++e) z[e] = 0.0f;
  return z;
}

// A fragment (16x32 bf16): row fixed per lane; VGPR j holds K pair
//   j<4 : K = 8*hi + 2j     ;  j>=4 : K = 16 + 8*hi + 2(j-4)
__device__ __forceinline__ v16bf load_fragA(const u16* base, int row, int stride,
                                            int k0, int hi) {
  FragAB f;
#pragma unroll
  for (int j = 0; j < 8; ++j) {
    int kk = k0 + ((j < 4) ? (8 * hi + 2 * j) : (16 + 8 * hi + 2 * (j - 4)));
    f.u[j] = *(const u32*)(base + row * stride + kk);
  }
  return f.v;
}
// B fragment (32x16 bf16): col fixed per lane; VGPR j holds K pair (same as A)
// LDS layout for the B operand is [col][k] so pairs are contiguous.
__device__ __forceinline__ v16bf load_fragB(const u16* base, int col, int stride,
                                            int k0, int hi) {
  FragAB f;
#pragma unroll
  for (int j = 0; j < 8; ++j) {
    int kk = k0 + ((j < 4) ? (8 * hi + 2 * j) : (16 + 8 * hi + 2 * (j - 4)));
    f.u[j] = *(const u32*)(base + col * stride + kk);
  }
  return f.v;
}
// Scale an A fragment (single row per lane) by a scalar.
__device__ __forceinline__ v16bf scale_frag(v16bf a, float f) {
  FragAB in, out;
  in.v = a;
#pragma unroll
  for (int j = 0; j < 8; ++j) {
    u32 w = in.u[j];
    float f0 = bf2f((u16)(w & 0xFFFFu)) * f;
    float f1 = bf2f((u16)(w >> 16)) * f;
    out.u[j] = (u32)f2bf(f0) | ((u32)f2bf(f1) << 16);
  }
  return out.v;
}

// ---------- TDM: async DMA of a 2D bf16 tile (rows x tile_d0) into LDS ----------
// D# per CDNA5 ISA ch.8: group0 = {count|flags, lds_addr, global_addr, type=2},
// group1 = {data_size=2B, tensor_dim0/1, tile_dim0/1, tensor_dim0_stride}.
// tensor_dim1 = remaining rows so out-of-range rows read as zero.
__device__ __forceinline__ void tdm_load_2d(u32 lds_off, const u16* gptr,
                                            u32 tensor_d0, u32 tensor_d1,
                                            u32 tile_d0, u32 tile_d1, u32 stride0) {
#if HAVE_TDM
  u64 ga = (u64)(size_t)gptr;
  v4u g0;
  g0[0] = 1u;                                   // count=1 valid user descriptor
  g0[1] = lds_off;                              // LDS byte address
  g0[2] = (u32)(ga & 0xFFFFFFFFull);            // global_addr[31:0]
  g0[3] = (u32)((ga >> 32) & 0x01FFFFFFull) | (2u << 30);  // addr[56:32], type=2
  u32 w[8];
  w[0] = 0x00010000u;                           // wg_mask=0, data_size=1 (2 bytes)
  w[1] = (tensor_d0 & 0xFFFFu) << 16;           // tensor_dim0[15:0]
  w[2] = ((tensor_d0 >> 16) & 0xFFFFu) | ((tensor_d1 & 0xFFFFu) << 16);
  w[3] = ((tensor_d1 >> 16) & 0xFFFFu) | (tile_d0 << 16);   // tile_dim0
  w[4] = tile_d1 & 0xFFFFu;                     // tile_dim1 (tile_dim2 = 0)
  w[5] = stride0;                               // tensor_dim0_stride[31:0]
  w[6] = 0;                                     // stride0 hi / stride1 lo
  w[7] = 0;
  v8i g1;
#pragma unroll
  for (int i = 0; i < 8; ++i) g1[i] = (int)w[i];
  v4i z4 = {0, 0, 0, 0};
#ifdef TDM_6ARG
  v8i z8 = {0, 0, 0, 0, 0, 0, 0, 0};
  __builtin_amdgcn_tensor_load_to_lds(g0, g1, z4, z4, z8, 0);
#else
  __builtin_amdgcn_tensor_load_to_lds(g0, g1, z4, z4, 0);
#endif
#else
  (void)lds_off; (void)gptr; (void)tensor_d0; (void)tensor_d1;
  (void)tile_d0; (void)tile_d1; (void)stride0;
#endif
}

// ---------- casts ----------
__global__ void cast_bf16_k(const float* __restrict__ src, u16* __restrict__ dst, int n) {
  int i = blockIdx.x * 256 + threadIdx.x;
  if (i < n) dst[i] = f2bf(src[i]);
}
// dst[n][k] = bf16(src[k][n])  (pre-transpose weights so GEMM B operand is K-contiguous)
__global__ void cast_bf16_t_k(const float* __restrict__ src, u16* __restrict__ dst,
                              int K, int N) {
  int i = blockIdx.x * 256 + threadIdx.x;
  if (i >= K * N) return;
  int n = i % N, k = i / N;
  dst[(size_t)n * K + k] = f2bf(src[(size_t)k * N + n]);
}

// ---------- bf16 WMMA GEMM: C[M,N] = A[M,K] @ Bt[N,K]^T, f32 out ----------
// block tile 128x128, 8 waves (4x2), wave tile 32x64 (2x4 fragments), K step 32.
// Tiles staged by the Tensor Data Mover (double buffered on TENSORcnt).
__global__ __launch_bounds__(256) void gemm_bf16_rrt(
    const u16* __restrict__ A, const u16* __restrict__ Bt, float* __restrict__ C,
    int M, int N, int K) {
  __shared__ u16 sA[2][128][32];   // [m][k]
  __shared__ u16 sB[2][128][32];   // [n][k]
  const int tid = threadIdx.x;
  const int wid = tid >> 5, lane = tid & 31;
  const int hi  = lane >> 4, lo = lane & 15;
  const int wM  = wid >> 1, wN = wid & 1;
  const int m0  = blockIdx.y * 128, n0 = blockIdx.x * 128;
  const int nIter = K >> 5;
  const u32 rowsB = (u32)(N - n0);

  v8f acc[2][4];
#pragma unroll
  for (int a = 0; a < 2; ++a)
#pragma unroll
    for (int b = 0; b < 4; ++b) acc[a][b] = vzero8();

#if HAVE_TDM
  if (wid == 0) {
    tdm_load_2d((u32)(size_t)&sA[0][0][0], A + (size_t)m0 * K,
                (u32)K, 128u, 32u, 128u, (u32)K);
    tdm_load_2d((u32)(size_t)&sB[0][0][0], Bt + (size_t)n0 * K,
                (u32)K, rowsB, 32u, 128u, (u32)K);
  }
#endif
  for (int it = 0; it < nIter; ++it) {
    const int cur = it & 1;
#if HAVE_TDM
    if (wid == 0) {
      if (it + 1 < nIter) {
        const int k0n = (it + 1) << 5;
        tdm_load_2d((u32)(size_t)&sA[cur ^ 1][0][0], A + (size_t)m0 * K + k0n,
                    (u32)(K - k0n), 128u, 32u, 128u, (u32)K);
        tdm_load_2d((u32)(size_t)&sB[cur ^ 1][0][0], Bt + (size_t)n0 * K + k0n,
                    (u32)(K - k0n), rowsB, 32u, 128u, (u32)K);
        __builtin_amdgcn_s_wait_tensorcnt(2);  // current buffer's pair done
      } else {
        __builtin_amdgcn_s_wait_tensorcnt(0);
      }
    }
#else
    const int k0 = it << 5;
    for (int i = tid; i < 512; i += 256) {           // A: 128x32 as uint4
      int m = i >> 2, kg = (i & 3) << 3;
      *(uint4*)&sA[cur][m][kg] = *(const uint4*)(A + (size_t)(m0 + m) * K + k0 + kg);
    }
    for (int i = tid; i < 512; i += 256) {           // B: 128x32 as uint4
      int n = i >> 2, kg = (i & 3) << 3;
      uint4 v = make_uint4(0u, 0u, 0u, 0u);
      if ((u32)n < rowsB) v = *(const uint4*)(Bt + (size_t)(n0 + n) * K + k0 + kg);
      *(uint4*)&sB[cur][n][kg] = v;
    }
#endif
    __syncthreads();
    v16bf af[2], bfr[4];
#pragma unroll
    for (int fm = 0; fm < 2; ++fm)
      af[fm] = load_fragA(&sA[cur][0][0], wM * 32 + fm * 16 + lo, 32, 0, hi);
#pragma unroll
    for (int fn = 0; fn < 4; ++fn)
      bfr[fn] = load_fragB(&sB[cur][0][0], wN * 64 + fn * 16 + lo, 32, 0, hi);
#pragma unroll
    for (int fm = 0; fm < 2; ++fm)
#pragma unroll
      for (int fn = 0; fn < 4; ++fn)
        acc[fm][fn] = __builtin_amdgcn_wmma_f32_16x16x32_bf16(
            false, af[fm], false, bfr[fn], (short)0, acc[fm][fn], false, false);
    __syncthreads();
  }
#pragma unroll
  for (int fm = 0; fm < 2; ++fm)
#pragma unroll
    for (int fn = 0; fn < 4; ++fn)
#pragma unroll
      for (int e = 0; e < 8; ++e) {
        int row = m0 + wM * 32 + fm * 16 + e + 8 * hi;
        int col = n0 + wN * 64 + fn * 16 + lo;
        if (col < N) C[(size_t)row * N + col] = acc[fm][fn][e];
      }
}

// ---------- dt = softplus(raw + bias); dA = dt * (-exp(A_log)) ----------
__global__ void dtprep_k(const float* __restrict__ zx, const float* __restrict__ dt_bias,
                         const float* __restrict__ A_log, float* __restrict__ dt,
                         float* __restrict__ dA) {
  int i = blockIdx.x * 256 + threadIdx.x;
  if (i >= TOK * NH) return;
  int t = i >> 5, h = i & 31;
  float raw = zx[(size_t)t * NPROJ + DIN + CONVD + h] + dt_bias[h];
  float sp  = (raw > 20.0f) ? raw : log1pf(__expf(raw));
  dt[i] = sp;
  dA[i] = -sp * __expf(A_log[h]);
}

// ---------- causal depthwise conv4 + SiLU; split into X, X*dt, B, C (bf16) ----------
__global__ void conv_k(const float* __restrict__ zx, const float* __restrict__ cw,
                       const float* __restrict__ cb, const float* __restrict__ dt,
                       u16* __restrict__ Xs, u16* __restrict__ Xdt,
                       u16* __restrict__ Bb, u16* __restrict__ Cb) {
  int i = blockIdx.x * 256 + threadIdx.x;
  if (i >= TOK * CONVD) return;
  int c = i % CONVD, t = i / CONVD;
  int l = t & (SEQL - 1);
  float acc = cb[c];
#pragma unroll
  for (int k = 0; k < 4; ++k) {
    int ll = l - 3 + k;
    if (ll >= 0) acc += cw[c * 4 + k] * zx[(size_t)(t - 3 + k) * NPROJ + DIN + c];
  }
  float v = siluf(acc);
  if (c < DIN) {
    int h = c >> 7;
    Xs[(size_t)t * DIN + c]  = f2bf(v);
    Xdt[(size_t)t * DIN + c] = f2bf(v * dt[t * NH + h]);
  } else if (c < DIN + DST) {
    Bb[t * DST + (c - DIN)] = f2bf(v);
  } else {
    Cb[t * DST + (c - DIN - DST)] = f2bf(v);
  }
}

// ---------- per (b,chunk,head): states[p][n] = sum_l Xdt[l][p]*decay[l]*B[l][n] ----------
__global__ __launch_bounds__(256) void states_k(
    const u16* __restrict__ Xdt, const u16* __restrict__ Bb, const float* __restrict__ dA,
    float* __restrict__ states, float* __restrict__ acs, float* __restrict__ csum) {
  __shared__ float s_cs[64];
  __shared__ u16 sXT[128][64];  // A operand [p][l], decay applied
  __shared__ u16 sBT[64][64];   // B operand [n][l]
  const int blk = blockIdx.x;   // (b*NCH + c)*NH + h
  const int h = blk & 31, bc = blk >> 5, c = bc & 63, b = bc >> 6;
  const int t0 = b * SEQL + c * CLEN;
  const int tid = threadIdx.x;

  if (tid == 0) {
    float run = 0.0f;
    for (int l = 0; l < 64; ++l) {
      run += dA[(t0 + l) * NH + h];
      s_cs[l] = run;
      acs[(size_t)(t0 + l) * NH + h] = run;
    }
    csum[blk] = run;
  }
  __syncthreads();
  const float cstot = s_cs[63];
  for (int i = tid; i < 64 * 16; i += 256) {      // Xdt^T with decay, uint4 loads
    int l = i >> 4, pg = (i & 15) << 3;
    Pk8 d;
    d.q = *(const uint4*)(Xdt + (size_t)(t0 + l) * DIN + h * HD + pg);
    float dec = __expf(cstot - s_cs[l]);
#pragma unroll
    for (int j = 0; j < 8; ++j) sXT[pg + j][l] = f2bf(bf2f(d.h[j]) * dec);
  }
  for (int i = tid; i < 64 * 8; i += 256) {       // B^T, uint4 loads
    int l = i >> 3, ng = (i & 7) << 3;
    Pk8 d;
    d.q = *(const uint4*)(Bb + (size_t)(t0 + l) * DST + ng);
#pragma unroll
    for (int j = 0; j < 8; ++j) sBT[ng + j][l] = d.h[j];
  }
  __syncthreads();

  const int wid = tid >> 5, lane = tid & 31, hi = lane >> 4, lo = lane & 15;
  const int wM = wid >> 1, wN = wid & 1;   // 4x2 waves -> 128x64 output
  v8f acc[2][2];
#pragma unroll
  for (int a = 0; a < 2; ++a)
#pragma unroll
    for (int d = 0; d < 2; ++d) acc[a][d] = vzero8();

#pragma unroll
  for (int k0 = 0; k0 < 64; k0 += 32) {
    v16bf af[2], bfr[2];
#pragma unroll
    for (int fm = 0; fm < 2; ++fm)
      af[fm] = load_fragA(&sXT[0][0], wM * 32 + fm * 16 + lo, 64, k0, hi);
#pragma unroll
    for (int fn = 0; fn < 2; ++fn)
      bfr[fn] = load_fragB(&sBT[0][0], wN * 32 + fn * 16 + lo, 64, k0, hi);
#pragma unroll
    for (int fm = 0; fm < 2; ++fm)
#pragma unroll
      for (int fn = 0; fn < 2; ++fn)
        acc[fm][fn] = __builtin_amdgcn_wmma_f32_16x16x32_bf16(
            false, af[fm], false, bfr[fn], (short)0, acc[fm][fn], false, false);
  }
  float* out = states + (size_t)blk * (HD * DST);
#pragma unroll
  for (int fm = 0; fm < 2; ++fm)
#pragma unroll
    for (int fn = 0; fn < 2; ++fn)
#pragma unroll
      for (int e = 0; e < 8; ++e) {
        int p = wM * 32 + fm * 16 + e + 8 * hi;
        int n = wN * 32 + fn * 16 + lo;
        out[p * DST + n] = acc[fm][fn][e];
      }
}

// ---------- inter-chunk scan: S_in[c] = exp(cs[c-1])*S_in[c-1] + S[c-1] ----------
__global__ void scan_k(const float* __restrict__ states, const float* __restrict__ csum,
                       float* __restrict__ sin_) {
  int i = blockIdx.x * 256 + threadIdx.x;  // NB*CHS
  if (i >= NB * CHS) return;
  int b = i / CHS, r = i % CHS;
  int h = r / (HD * DST);
  size_t base = (size_t)b * NCH * CHS + r;
  float s = 0.0f;
  for (int c = 0; c < NCH; ++c) {
    if (c + 2 < NCH) __builtin_prefetch(&states[base + (size_t)(c + 2) * CHS], 0, 0);
    sin_[base + (size_t)c * CHS] = s;
    s = __expf(csum[(b * NCH + c) * NH + h]) * s + states[base + (size_t)c * CHS];
  }
}

// ---------- per (b,chunk,head): Y = (L∘(C·Bᵀ))@Xdt + (C·decay)@S_inᵀ + D·X ----------
__global__ __launch_bounds__(256) void yout_k(
    const u16* __restrict__ Cb, const u16* __restrict__ Bb, const u16* __restrict__ Xdt,
    const u16* __restrict__ Xs, const float* __restrict__ sin_,
    const float* __restrict__ acs, const float* __restrict__ Dp, float* __restrict__ Y) {
  __shared__ float s_cs[64];
  __shared__ u16 sC[64][64];    // C chunk [l][n]     (A op, GEMM1/GEMM3)
  __shared__ u16 sBM[64][64];   // Bc [s][n], then Mmat [l][s]
  __shared__ u16 sXT[128][64];  // Xdt^T [p][s]       (B op, GEMM2)
  __shared__ u16 sS[128][64];   // S_in [p][n]        (B op, GEMM3)
  const int blk = blockIdx.x;
  const int h = blk & 31, bc = blk >> 5, c = bc & 63, b = bc >> 6;
  const int t0 = b * SEQL + c * CLEN;
  const int tid = threadIdx.x;
  const int wid = tid >> 5, lane = tid & 31, hi = lane >> 4, lo = lane & 15;

#if HAVE_TDM
  // both chunk tiles are contiguous 64*64 bf16 blocks -> 1D TDM copies
  if (wid == 0) {
    tdm_load_2d((u32)(size_t)&sC[0][0],  Cb + (size_t)t0 * DST, 4096u, 1u, 4096u, 1u, 4096u);
    tdm_load_2d((u32)(size_t)&sBM[0][0], Bb + (size_t)t0 * DST, 4096u, 1u, 4096u, 1u, 4096u);
  }
#else
  for (int i = tid; i < 64 * 8; i += 256) {
    int l = i >> 3, ng = (i & 7) << 3;
    *(uint4*)&sC[l][ng]  = *(const uint4*)(Cb + (size_t)(t0 + l) * DST + ng);
    *(uint4*)&sBM[l][ng] = *(const uint4*)(Bb + (size_t)(t0 + l) * DST + ng);
  }
#endif
  if (tid < 64) s_cs[tid] = acs[(size_t)(t0 + tid) * NH + h];
  for (int i = tid; i < 64 * 16; i += 256) {      // Xdt^T, uint4 loads
    int l = i >> 4, pg = (i & 15) << 3;
    Pk8 d;
    d.q = *(const uint4*)(Xdt + (size_t)(t0 + l) * DIN + h * HD + pg);
#pragma unroll
    for (int j = 0; j < 8; ++j) sXT[pg + j][l] = d.h[j];
  }
  for (int i = tid; i < 128 * 16; i += 256) {     // S_in f32 -> bf16, float4 loads
    int p = i >> 4, ng = (i & 15) << 2;
    Pk4f d;
    d.q = *(const float4*)(sin_ + (size_t)blk * (HD * DST) + p * DST + ng);
    *(u32*)&sS[p][ng]     = (u32)f2bf(d.f[0]) | ((u32)f2bf(d.f[1]) << 16);
    *(u32*)&sS[p][ng + 2] = (u32)f2bf(d.f[2]) | ((u32)f2bf(d.f[3]) << 16);
  }
#if HAVE_TDM
  if (wid == 0) __builtin_amdgcn_s_wait_tensorcnt(0);
#endif
  __syncthreads();

  // GEMM1 (waves 0..3): G[l][s] = sum_n C[l][n] * B[s][n]
  v8f g[2][2];
#pragma unroll
  for (int a = 0; a < 2; ++a)
#pragma unroll
    for (int d = 0; d < 2; ++d) g[a][d] = vzero8();
  if (wid < 4) {
    const int gM = wid >> 1, gN = wid & 1;
#pragma unroll
    for (int k0 = 0; k0 < 64; k0 += 32) {
      v16bf af[2], bfr[2];
#pragma unroll
      for (int fm = 0; fm < 2; ++fm)
        af[fm] = load_fragA(&sC[0][0], gM * 32 + fm * 16 + lo, 64, k0, hi);
#pragma unroll
      for (int fn = 0; fn < 2; ++fn)
        bfr[fn] = load_fragB(&sBM[0][0], gN * 32 + fn * 16 + lo, 64, k0, hi);
#pragma unroll
      for (int fm = 0; fm < 2; ++fm)
#pragma unroll
        for (int fn = 0; fn < 2; ++fn)
          g[fm][fn] = __builtin_amdgcn_wmma_f32_16x16x32_bf16(
              false, af[fm], false, bfr[fn], (short)0, g[fm][fn], false, false);
    }
  }
  __syncthreads();
  // apply segsum mask + decay, overwrite sBM with Mmat (bf16, row-major [l][s])
  if (wid < 4) {
    const int gM = wid >> 1, gN = wid & 1;
#pragma unroll
    for (int fm = 0; fm < 2; ++fm)
#pragma unroll
      for (int fn = 0; fn < 2; ++fn)
#pragma unroll
        for (int e = 0; e < 8; ++e) {
          int l = gM * 32 + fm * 16 + e + 8 * hi;
          int s = gN * 32 + fn * 16 + lo;
          float f = (s <= l) ? __expf(s_cs[l] - s_cs[s]) : 0.0f;
          sBM[l][s] = f2bf(g[fm][fn][e] * f);
        }
  }
  __syncthreads();

  // GEMM2 + GEMM3 (all 8 waves, 2x4 -> 64x128)
  const int wM = wid >> 2, wN = wid & 3;
  v8f y[2][2];
#pragma unroll
  for (int a = 0; a < 2; ++a)
#pragma unroll
    for (int d = 0; d < 2; ++d) y[a][d] = vzero8();

#pragma unroll
  for (int k0 = 0; k0 < 64; k0 += 32) {  // Y += Mmat @ Xdt
    v16bf af[2], bfr[2];
#pragma unroll
    for (int fm = 0; fm < 2; ++fm)
      af[fm] = load_fragA(&sBM[0][0], wM * 32 + fm * 16 + lo, 64, k0, hi);
#pragma unroll
    for (int fn = 0; fn < 2; ++fn)
      bfr[fn] = load_fragB(&sXT[0][0], wN * 32 + fn * 16 + lo, 64, k0, hi);
#pragma unroll
    for (int fm = 0; fm < 2; ++fm)
#pragma unroll
      for (int fn = 0; fn < 2; ++fn)
        y[fm][fn] = __builtin_amdgcn_wmma_f32_16x16x32_bf16(
            false, af[fm], false, bfr[fn], (short)0, y[fm][fn], false, false);
  }
  float fac0 = __expf(s_cs[wM * 32 + 0 * 16 + lo]);
  float fac1 = __expf(s_cs[wM * 32 + 1 * 16 + lo]);
#pragma unroll
  for (int k0 = 0; k0 < 64; k0 += 32) {  // Y += (C*decay_out) @ S_in^T
    v16bf af[2], bfr[2];
    af[0] = scale_frag(load_fragA(&sC[0][0], wM * 32 + 0 * 16 + lo, 64, k0, hi), fac0);
    af[1] = scale_frag(load_fragA(&sC[0][0], wM * 32 + 1 * 16 + lo, 64, k0, hi), fac1);
#pragma unroll
    for (int fn = 0; fn < 2; ++fn)
      bfr[fn] = load_fragB(&sS[0][0], wN * 32 + fn * 16 + lo, 64, k0, hi);
#pragma unroll
    for (int fm = 0; fm < 2; ++fm)
#pragma unroll
      for (int fn = 0; fn < 2; ++fn)
        y[fm][fn] = __builtin_amdgcn_wmma_f32_16x16x32_bf16(
            false, af[fm], false, bfr[fn], (short)0, y[fm][fn], false, false);
  }
  const float Dh = Dp[h];
#pragma unroll
  for (int fm = 0; fm < 2; ++fm)
#pragma unroll
    for (int fn = 0; fn < 2; ++fn)
#pragma unroll
      for (int e = 0; e < 8; ++e) {
        int l = wM * 32 + fm * 16 + e + 8 * hi;
        int p = wN * 32 + fn * 16 + lo;
        size_t idx = (size_t)(t0 + l) * DIN + h * HD + p;
        Y[idx] = y[fm][fn][e] + Dh * bf2f(Xs[idx]);
      }
}

// ---------- gated RMSNorm: gy = bf16( Y*silu(z) * rsqrt(mean(g^2)+eps) * w ) ----------
__global__ __launch_bounds__(256) void gate_k(const float* __restrict__ Y,
                                              const float* __restrict__ zx,
                                              const float* __restrict__ nw,
                                              u16* __restrict__ gy) {
  __shared__ float red[256];
  const int t = blockIdx.x, tid = threadIdx.x;
  float g[16];
  float ss = 0.0f;
#pragma unroll
  for (int i = 0; i < 16; ++i) {
    int cidx = tid + i * 256;
    float z  = zx[(size_t)t * NPROJ + cidx];
    float gg = Y[(size_t)t * DIN + cidx] * siluf(z);
    g[i] = gg;
    ss += gg * gg;
  }
  red[tid] = ss;
  __syncthreads();
  for (int s = 128; s > 0; s >>= 1) {
    if (tid < s) red[tid] += red[tid + s];
    __syncthreads();
  }
  float rms = rsqrtf(red[0] / (float)DIN + 1e-5f);
#pragma unroll
  for (int i = 0; i < 16; ++i) {
    int cidx = tid + i * 256;
    gy[(size_t)t * DIN + cidx] = f2bf(g[i] * rms * nw[cidx]);
  }
}

// ======================== host launcher ========================
extern "C" void kernel_launch(void* const* d_in, const int* in_sizes, int n_in,
                              void* d_out, int out_size, void* d_ws, size_t ws_size,
                              hipStream_t stream) {
  (void)in_sizes; (void)n_in; (void)out_size; (void)ws_size;
  const float* x       = (const float*)d_in[0];
  const float* W_in    = (const float*)d_in[1];
  const float* conv_w  = (const float*)d_in[2];
  const float* conv_b  = (const float*)d_in[3];
  const float* dt_bias = (const float*)d_in[4];
  const float* A_log   = (const float*)d_in[5];
  const float* Dp      = (const float*)d_in[6];
  const float* norm_w  = (const float*)d_in[7];
  const float* W_out   = (const float*)d_in[8];
  float* out = (float*)d_out;

  char* ws = (char*)d_ws;
  size_t off = 0;
  auto alloc = [&](size_t bytes) -> void* {
    void* p = ws + off;
    off = (off + bytes + 255) & ~(size_t)255;
    return p;
  };
  u16*   xb     = (u16*)  alloc((size_t)TOK * DM * 2);
  u16*   WibT   = (u16*)  alloc((size_t)DM * NPROJ * 2);   // [NPROJ][DM]
  u16*   WobT   = (u16*)  alloc((size_t)DIN * DM * 2);     // [DM][DIN]
  float* zx     = (float*)alloc((size_t)TOK * NPROJ * 4);
  float* dtv    = (float*)alloc((size_t)TOK * NH * 4);
  float* dAv    = (float*)alloc((size_t)TOK * NH * 4);
  float* acs    = (float*)alloc((size_t)TOK * NH * 4);
  u16*   Xs     = (u16*)  alloc((size_t)TOK * DIN * 2);
  u16*   Xdt    = (u16*)  alloc((size_t)TOK * DIN * 2);
  u16*   Bb     = (u16*)  alloc((size_t)TOK * DST * 2);
  u16*   Cbuf   = (u16*)  alloc((size_t)TOK * DST * 2);
  float* csum   = (float*)alloc((size_t)NB * NCH * NH * 4);
  float* states = (float*)alloc((size_t)NB * NCH * CHS * 4);
  float* sinb   = (float*)alloc((size_t)NB * NCH * CHS * 4);
  float* Yb     = (float*)alloc((size_t)TOK * DIN * 4);
  u16*   gy     = (u16*)  alloc((size_t)TOK * DIN * 2);

  // 1) casts (weights transposed so GEMM B operand is K-contiguous)
  cast_bf16_k<<<(TOK * DM + 255) / 256, 256, 0, stream>>>(x, xb, TOK * DM);
  cast_bf16_t_k<<<(DM * NPROJ + 255) / 256, 256, 0, stream>>>(W_in, WibT, DM, NPROJ);
  cast_bf16_t_k<<<(DIN * DM + 255) / 256, 256, 0, stream>>>(W_out, WobT, DIN, DM);

  // 2) in_proj GEMM: zx[TOK, NPROJ] = x @ W_in
  gemm_bf16_rrt<<<dim3((NPROJ + 127) / 128, TOK / 128), 256, 0, stream>>>(
      xb, WibT, zx, TOK, NPROJ, DM);

  // 3) dt softplus + dA
  dtprep_k<<<(TOK * NH + 255) / 256, 256, 0, stream>>>(zx, dt_bias, A_log, dtv, dAv);

  // 4) conv + SiLU + split
  conv_k<<<(TOK * CONVD + 255) / 256, 256, 0, stream>>>(
      zx, conv_w, conv_b, dtv, Xs, Xdt, Bb, Cbuf);

  // 5) per-chunk states (WMMA)
  states_k<<<NB * NCH * NH, 256, 0, stream>>>(Xdt, Bb, dAv, states, acs, csum);

  // 6) inter-chunk scan
  scan_k<<<(NB * CHS + 255) / 256, 256, 0, stream>>>(states, csum, sinb);

  // 7) Y = diag + offdiag + skip (WMMA)
  yout_k<<<NB * NCH * NH, 256, 0, stream>>>(Cbuf, Bb, Xdt, Xs, sinb, acs, Dp, Yb);

  // 8) gated RMSNorm -> bf16
  gate_k<<<TOK, 256, 0, stream>>>(Yb, zx, norm_w, gy);

  // 9) out_proj GEMM -> f32 output
  gemm_bf16_rrt<<<dim3((DM + 127) / 128, TOK / 128), 256, 0, stream>>>(
      gy, WobT, out, TOK, DM, DIN);
}
